// PointNetFeaturePropagation_68496138437067
// MI455X (gfx1250) — matrix operand order repaired and checked by reading, not compiled, not run on hardware
//
#include <hip/hip_runtime.h>
#include <hip/hip_bf16.h>

// ---------------------------------------------------------------------------
// Problem constants (from reference)
// ---------------------------------------------------------------------------
constexpr int kB   = 8;
constexpr int kN   = 4096;
constexpr int kS   = 1024;
constexpr int kD1  = 128;
constexpr int kD2  = 256;
constexpr int kINC = 384;           // D1 + D2
constexpr int kH   = 256;           // MLP width (both layers)
constexpr int kM   = kB * kN;       // 32768 rows in the flattened GEMM
constexpr float kEps = 1e-5f;

typedef __attribute__((ext_vector_type(16))) __bf16       v16bf;
typedef __attribute__((ext_vector_type(8)))  float        v8f;
typedef __attribute__((ext_vector_type(4)))  unsigned int u32x4;

union FragAB { v16bf v; u32x4 u[2]; };
union Pack8  { unsigned short h[8]; u32x4 v; };

__device__ __forceinline__ unsigned short f2bf(float x) {
  // round-to-nearest-even fp32 -> bf16 (bit pattern only; stored as u16)
  unsigned u = __float_as_uint(x);
  u += 0x7FFFu + ((u >> 16) & 1u);
  return (unsigned short)(u >> 16);
}

// ---------------------------------------------------------------------------
// Weight fp32 -> bf16
// ---------------------------------------------------------------------------
__global__ __launch_bounds__(256)
void f32_to_bf16_kernel(const float* __restrict__ in,
                        unsigned short* __restrict__ out, int n) {
  int i = blockIdx.x * 256 + threadIdx.x;
  if (i < n) out[i] = f2bf(in[i]);
}

// ---------------------------------------------------------------------------
// 3-NN inverse-distance interpolation + concat -> bf16 activation matrix
// One thread per (b, n). Block = 256 threads = 256 query points of batch b.
// Builds XP, XC : [M=32768, K=384] row-major bf16.
// ---------------------------------------------------------------------------
__global__ __launch_bounds__(256)
void interp_build_kernel(const float* __restrict__ xyz1,
                         const float* __restrict__ xyz2,
                         const float* __restrict__ p1,
                         const float* __restrict__ p2,
                         const float* __restrict__ c1,
                         const float* __restrict__ c2,
                         unsigned short* __restrict__ XP,
                         unsigned short* __restrict__ XC) {
  __shared__ float sx[kS], sy[kS], sz[kS];
  const int b  = blockIdx.x >> 4;          // 16 blocks per batch
  const int n0 = (blockIdx.x & 15) * 256;
  const int t  = threadIdx.x;

  const float* src = xyz2 + (size_t)b * 3 * kS;
  for (int i = t; i < kS; i += 256) {
    sx[i] = src[i];
    sy[i] = src[kS + i];
    sz[i] = src[2 * kS + i];
  }
  __syncthreads();

  const int n = n0 + t;
  const size_t xb = (size_t)b * 3 * kN;
  const float px = xyz1[xb + n];
  const float py = xyz1[xb + kN + n];
  const float pz = xyz1[xb + 2 * kN + n];

  float d0 = 1e30f, d1 = 1e30f, d2 = 1e30f;
  int   i0 = 0,     i1 = 0,     i2 = 0;
  for (int s = 0; s < kS; ++s) {
    float dx = px - sx[s], dy = py - sy[s], dz = pz - sz[s];
    float d = dx * dx + dy * dy + dz * dz;
    if (d < d2) {
      if (d < d1) {
        if (d < d0) { d2 = d1; i2 = i1; d1 = d0; i1 = i0; d0 = d; i0 = s; }
        else        { d2 = d1; i2 = i1; d1 = d;  i1 = s; }
      } else        { d2 = d;  i2 = s; }
    }
  }
  const float r0 = 1.f / (d0 + 1e-8f);
  const float r1 = 1.f / (d1 + 1e-8f);
  const float r2 = 1.f / (d2 + 1e-8f);
  const float rs = 1.f / (r0 + r1 + r2);
  const float w0 = r0 * rs, w1 = r1 * rs, w2 = r2 * rs;

  const size_t m = (size_t)b * kN + n;
  Pack8 pk;

  // cols [0,128): direct features (transpose read: coalesced across lanes)
  {
    const size_t pb = (size_t)b * kD1 * kN + n;   // p1[(b*D1+c)*N + n]
    for (int c0 = 0; c0 < kD1; c0 += 8) {
      #pragma unroll
      for (int e = 0; e < 8; ++e) pk.h[e] = f2bf(p1[pb + (size_t)(c0 + e) * kN]);
      *(u32x4*)&XP[m * kINC + c0] = pk.v;
      #pragma unroll
      for (int e = 0; e < 8; ++e) pk.h[e] = f2bf(c1[pb + (size_t)(c0 + e) * kN]);
      *(u32x4*)&XC[m * kINC + c0] = pk.v;
    }
  }
  // cols [128,384): interpolated features (gather)
  {
    const size_t fb = (size_t)b * kD2 * kS;       // f2[(b*D2+c)*S + idx]
    for (int c0 = 0; c0 < kD2; c0 += 8) {
      #pragma unroll
      for (int e = 0; e < 8; ++e) {
        const float* f = p2 + fb + (size_t)(c0 + e) * kS;
        pk.h[e] = f2bf(w0 * f[i0] + w1 * f[i1] + w2 * f[i2]);
      }
      *(u32x4*)&XP[m * kINC + kD1 + c0] = pk.v;
      #pragma unroll
      for (int e = 0; e < 8; ++e) {
        const float* f = c2 + fb + (size_t)(c0 + e) * kS;
        pk.h[e] = f2bf(w0 * f[i0] + w1 * f[i1] + w2 * f[i2]);
      }
      *(u32x4*)&XC[m * kINC + kD1 + c0] = pk.v;
    }
  }
}

// ---------------------------------------------------------------------------
// WMMA GEMM: Y[M,256] = X[M,K](bf16) @ W[256,K]^T(bf16) + bias, fp32 out.
// Wave tile 32Mx64N (2x4 x v_wmma_f32_16x16x32_bf16); 8 waves -> 64Mx256N.
// Fragment loads follow CDNA5 ISA 16-bit A/B VGPR layouts.
// ---------------------------------------------------------------------------
template <int KDIM>
__global__ __launch_bounds__(256)
void gemm_bf16_bias_kernel(const unsigned short* __restrict__ X,
                           const unsigned short* __restrict__ W,
                           const float* __restrict__ bias,
                           float* __restrict__ Y) {
  constexpr int NOUT = 256;
  const int lane = threadIdx.x & 31;
  const int wave = threadIdx.x >> 5;
  const int hf   = lane >> 4;        // lane half: selects K sub-ranges
  const int l16  = lane & 15;
  const int mBase = blockIdx.x * 64 + (wave >> 2) * 32;
  const int nBase = (wave & 3) * 64;

  v8f acc[2][4] = {};

  const unsigned short* aRow0 = X + (size_t)(mBase + l16) * KDIM;
  const unsigned short* aRow1 = X + (size_t)(mBase + 16 + l16) * KDIM;

  #pragma unroll
  for (int k0 = 0; k0 < KDIM; k0 += 32) {
    if (k0 + 32 < KDIM) {
      __builtin_prefetch(aRow0 + k0 + 32, 0, 0);
      __builtin_prefetch(aRow1 + k0 + 32, 0, 0);
    }
    FragAB a[2];
    // A 16x32 bf16: lanes 0-15 K={0..7,16..23}, lanes 16-31 K={8..15,24..31}
    a[0].u[0] = *(const u32x4*)(aRow0 + k0 + 8 * hf);
    a[0].u[1] = *(const u32x4*)(aRow0 + k0 + 8 * hf + 16);
    a[1].u[0] = *(const u32x4*)(aRow1 + k0 + 8 * hf);
    a[1].u[1] = *(const u32x4*)(aRow1 + k0 + 8 * hf + 16);

    FragAB bf[4];
    #pragma unroll
    for (int j = 0; j < 4; ++j) {
      // B 32x16 bf16: lane holds column n=l16; lanes 0-15 K={0..7,16..23}, lanes 16-31 K={8..15,24..31}
      const unsigned short* wRow =
          W + (size_t)(nBase + j * 16 + l16) * KDIM + k0 + 8 * hf;
      bf[j].u[0] = *(const u32x4*)(wRow);
      bf[j].u[1] = *(const u32x4*)(wRow + 16);
    }

    #pragma unroll
    for (int i = 0; i < 2; ++i)
      #pragma unroll
      for (int j = 0; j < 4; ++j)
        acc[i][j] = __builtin_amdgcn_wmma_f32_16x16x32_bf16(
            false, a[i].v, false, bf[j].v, (short)0, acc[i][j], false, false);
  }

  // C/D layout: VGPR r -> row m0 + r + 8*hf, col n0 + l16
  #pragma unroll
  for (int j = 0; j < 4; ++j) {
    const int col = nBase + j * 16 + l16;
    const float bv = bias[col];
    #pragma unroll
    for (int i = 0; i < 2; ++i) {
      #pragma unroll
      for (int r = 0; r < 8; ++r) {
        const int row = mBase + i * 16 + 8 * hf + r;
        Y[(size_t)row * NOUT + col] = acc[i][j][r] + bv;
      }
    }
  }
}

// ---------------------------------------------------------------------------
// Deterministic BN batch-stats: two-stage column reduction over M=32768 rows.
// ---------------------------------------------------------------------------
__global__ __launch_bounds__(256)
void col_partial_kernel(const float* __restrict__ Y, float* __restrict__ part) {
  const int c   = threadIdx.x;          // channel 0..255
  const int blk = blockIdx.x;           // 64 blocks x 512 rows
  const float* row = Y + (size_t)blk * 512 * kH;
  float s = 0.f, s2 = 0.f;
  for (int r = 0; r < 512; ++r) {
    float v = row[(size_t)r * kH + c];  // consecutive lanes -> coalesced
    s += v; s2 += v * v;
  }
  part[blk * 512 + c]       = s;
  part[blk * 512 + 256 + c] = s2;
}

__global__ __launch_bounds__(256)
void col_finalize_kernel(const float* __restrict__ part, float* __restrict__ stats) {
  const int c = threadIdx.x;
  float s = 0.f, s2 = 0.f;
  for (int b = 0; b < 64; ++b) {
    s  += part[b * 512 + c];
    s2 += part[b * 512 + 256 + c];
  }
  stats[c]       = s;
  stats[256 + c] = s2;
}

// ---------------------------------------------------------------------------
// BN + ReLU -> bf16 activation for next layer (elementwise, coalesced).
// ---------------------------------------------------------------------------
__global__ __launch_bounds__(256)
void bn_relu_bf16_kernel(const float* __restrict__ Y,
                         const float* __restrict__ stats,
                         const float* __restrict__ g,
                         const float* __restrict__ be,
                         unsigned short* __restrict__ X2) {
  const size_t i = (size_t)blockIdx.x * 256 + threadIdx.x;
  const int c = (int)(i & (kH - 1));
  const float invM = 1.f / (float)kM;
  const float mean = stats[c] * invM;
  const float var  = stats[kH + c] * invM - mean * mean;
  const float inv  = rsqrtf(var + kEps);
  float v = (Y[i] - mean) * inv * g[c] + be[c];
  X2[i] = f2bf(fmaxf(v, 0.f));
}

// ---------------------------------------------------------------------------
// Final BN + ReLU + transpose to [B, 256, N] via padded LDS tile.
// Grid: (N/32, 256/32, B); block 256 threads.
// ---------------------------------------------------------------------------
__global__ __launch_bounds__(256)
void bn_relu_transpose_kernel(const float* __restrict__ Y,
                              const float* __restrict__ stats,
                              const float* __restrict__ g,
                              const float* __restrict__ be,
                              float* __restrict__ out) {
  __shared__ float tile[32][33];
  const int n0 = blockIdx.x * 32;
  const int c0 = blockIdx.y * 32;
  const int b  = blockIdx.z;
  const int tc = threadIdx.x & 31;
  const int tr = threadIdx.x >> 5;     // 0..7
  const float invM = 1.f / (float)kM;

  const int c = c0 + tc;
  const float mean = stats[c] * invM;
  const float var  = stats[kH + c] * invM - mean * mean;
  const float inv  = rsqrtf(var + kEps);
  const float gc = g[c], bec = be[c];

  #pragma unroll
  for (int i = 0; i < 4; ++i) {
    const int nl = tr + i * 8;
    float v = Y[((size_t)(b * kN + n0 + nl)) * kH + c];
    v = (v - mean) * inv * gc + bec;
    tile[nl][tc] = fmaxf(v, 0.f);
  }
  __syncthreads();
  #pragma unroll
  for (int i = 0; i < 4; ++i) {
    const int cl = tr + i * 8;
    out[((size_t)(b * kH + c0 + cl)) * kN + n0 + tc] = tile[tc][cl];
  }
}

// ---------------------------------------------------------------------------
// Host launch
// ---------------------------------------------------------------------------
extern "C" void kernel_launch(void* const* d_in, const int* in_sizes, int n_in,
                              void* d_out, int out_size, void* d_ws, size_t ws_size,
                              hipStream_t stream) {
  (void)in_sizes; (void)n_in; (void)out_size; (void)ws_size;

  const float* xyz1 = (const float*)d_in[0];
  const float* xyz2 = (const float*)d_in[1];
  const float* p1   = (const float*)d_in[2];
  const float* p2   = (const float*)d_in[3];
  const float* c1   = (const float*)d_in[4];
  const float* c2   = (const float*)d_in[5];
  const float* Wp1  = (const float*)d_in[6];
  const float* bp1  = (const float*)d_in[7];
  const float* gp1  = (const float*)d_in[8];
  const float* bep1 = (const float*)d_in[9];
  const float* Wc1  = (const float*)d_in[10];
  const float* bc1  = (const float*)d_in[11];
  const float* gc1  = (const float*)d_in[12];
  const float* bec1 = (const float*)d_in[13];
  const float* Wp2  = (const float*)d_in[14];
  const float* bp2  = (const float*)d_in[15];
  const float* gp2  = (const float*)d_in[16];
  const float* bep2 = (const float*)d_in[17];
  const float* Wc2  = (const float*)d_in[18];
  const float* bc2  = (const float*)d_in[19];
  const float* gc2  = (const float*)d_in[20];
  const float* bec2 = (const float*)d_in[21];

  float* outP = (float*)d_out;
  float* outC = outP + (size_t)kB * kH * kN;

  // ---- workspace carve-up (256B-aligned sizes) ----
  char* ws = (char*)d_ws;
  size_t off = 0;
  const size_t SZ_X  = (size_t)kM * kINC * 2;   // 25.2 MB bf16 activations
  const size_t SZ_X2 = (size_t)kM * kH * 2;     // 16.8 MB bf16 layer-2 input
  const size_t SZ_Y  = (size_t)kM * kH * 4;     // 33.5 MB fp32 GEMM out
  unsigned short* XP  = (unsigned short*)(ws + off); off += SZ_X;
  unsigned short* XC  = (unsigned short*)(ws + off); off += SZ_X;
  unsigned short* X2P = (unsigned short*)(ws + off); off += SZ_X2;
  unsigned short* X2C = (unsigned short*)(ws + off); off += SZ_X2;
  float* YP = (float*)(ws + off); off += SZ_Y;  // reused layer1 & layer2
  float* YC = (float*)(ws + off); off += SZ_Y;
  unsigned short* WP1b = (unsigned short*)(ws + off); off += (size_t)kH * kINC * 2;
  unsigned short* WC1b = (unsigned short*)(ws + off); off += (size_t)kH * kINC * 2;
  unsigned short* WP2b = (unsigned short*)(ws + off); off += (size_t)kH * kH * 2;
  unsigned short* WC2b = (unsigned short*)(ws + off); off += (size_t)kH * kH * 2;
  float* PART = (float*)(ws + off); off += (size_t)64 * 512 * 4;
  float* STP1 = (float*)(ws + off); off += 512 * 4;
  float* STC1 = (float*)(ws + off); off += 512 * 4;
  float* STP2 = (float*)(ws + off); off += 512 * 4;
  float* STC2 = (float*)(ws + off); off += 512 * 4;

  // ---- weights -> bf16 ----
  f32_to_bf16_kernel<<<(kH * kINC + 255) / 256, 256, 0, stream>>>(Wp1, WP1b, kH * kINC);
  f32_to_bf16_kernel<<<(kH * kINC + 255) / 256, 256, 0, stream>>>(Wc1, WC1b, kH * kINC);
  f32_to_bf16_kernel<<<(kH * kH + 255) / 256, 256, 0, stream>>>(Wp2, WP2b, kH * kH);
  f32_to_bf16_kernel<<<(kH * kH + 255) / 256, 256, 0, stream>>>(Wc2, WC2b, kH * kH);

  // ---- 3-NN interpolation + concat -> bf16 activations ----
  interp_build_kernel<<<kB * (kN / 256), 256, 0, stream>>>(
      xyz1, xyz2, p1, p2, c1, c2, XP, XC);

  // ---- layer 1 GEMMs (K=384) ----
  gemm_bf16_bias_kernel<kINC><<<kM / 64, 256, 0, stream>>>(XP, WP1b, bp1, YP);
  gemm_bf16_bias_kernel<kINC><<<kM / 64, 256, 0, stream>>>(XC, WC1b, bc1, YC);

  // ---- BN stats + BN/ReLU -> bf16 ----
  col_partial_kernel<<<64, 256, 0, stream>>>(YP, PART);
  col_finalize_kernel<<<1, 256, 0, stream>>>(PART, STP1);
  bn_relu_bf16_kernel<<<(kM * kH) / 256, 256, 0, stream>>>(YP, STP1, gp1, bep1, X2P);
  col_partial_kernel<<<64, 256, 0, stream>>>(YC, PART);
  col_finalize_kernel<<<1, 256, 0, stream>>>(PART, STC1);
  bn_relu_bf16_kernel<<<(kM * kH) / 256, 256, 0, stream>>>(YC, STC1, gc1, bec1, X2C);

  // ---- layer 2 GEMMs (K=256) ----
  gemm_bf16_bias_kernel<kH><<<kM / 64, 256, 0, stream>>>(X2P, WP2b, bp2, YP);
  gemm_bf16_bias_kernel<kH><<<kM / 64, 256, 0, stream>>>(X2C, WC2b, bc2, YC);

  // ---- BN stats + BN/ReLU + transpose -> outputs ----
  dim3 tgrid(kN / 32, kH / 32, kB);
  col_partial_kernel<<<64, 256, 0, stream>>>(YP, PART);
  col_finalize_kernel<<<1, 256, 0, stream>>>(PART, STP2);
  bn_relu_transpose_kernel<<<tgrid, 256, 0, stream>>>(YP, STP2, gp2, bep2, outP);
  col_partial_kernel<<<64, 256, 0, stream>>>(YC, PART);
  col_finalize_kernel<<<1, 256, 0, stream>>>(PART, STC2);
  bn_relu_transpose_kernel<<<tgrid, 256, 0, stream>>>(YC, STC2, gc2, bec2, outC);
}
